// DotsVisionAttention_60902636257901
// MI455X (gfx1250) — matrix operand order repaired and checked by reading, not compiled, not run on hardware
//
#include <hip/hip_runtime.h>
#include <hip/hip_bf16.h>

typedef __attribute__((ext_vector_type(16))) __bf16 v16bf;
typedef __attribute__((ext_vector_type(8)))  __bf16 v8bf;
typedef __attribute__((ext_vector_type(8)))  float  v8f;

#define S_TOK   8192
#define DIM     1280
#define NH      16
#define HD      80
#define HDP     96      // head_dim padded to 3 WMMA K-steps of 32
#define NSEG    8
#define SEGL    1024
#define NQKV    3840

// ---------------------------------------------------------------------------
// CDNA5 async global->LDS copy (ASYNCcnt-tracked, per-lane LDS destination).
// Generic pointers to __shared__ have the LDS byte offset in their low 32 bits
// (flat LDS aperture).
// ---------------------------------------------------------------------------
__device__ __forceinline__ void async_ld16(void* lds, const void* gptr) {
    unsigned l = (unsigned)(unsigned long long)lds;
    asm volatile("global_load_async_to_lds_b128 %0, %1, off"
                 :: "v"(l), "v"(gptr) : "memory");
}
__device__ __forceinline__ void wait_async0() {
    asm volatile("s_wait_asynccnt 0x0" ::: "memory");
}

// ---------------------------------------------------------------------------
// f32 -> bf16 conversion
// ---------------------------------------------------------------------------
__global__ void cvt_bf16_kernel(const float* __restrict__ in,
                                __bf16* __restrict__ out, int n) {
    int i = blockIdx.x * 256 + threadIdx.x;
    int stride = gridDim.x * 256;
    for (; i < n; i += stride) out[i] = (__bf16)in[i];
}

// ---------------------------------------------------------------------------
// Tiled bf16 GEMM:  out[M,N] = A[M,K] @ B[N,K]^T + bias
// Block tile 128x256, K-step 32, double-buffered async LDS staging.
// 8 waves (2x4), wave tile 64x64 -> 16 v_wmma per wave per K-step.
// ---------------------------------------------------------------------------
template <int OUTF32>
__global__ __launch_bounds__(256) void gemm_bf16_kernel(
    const __bf16* __restrict__ A, const __bf16* __restrict__ B,
    const float* __restrict__ bias, __bf16* __restrict__ outb,
    float* __restrict__ outf, int M, int N, int K) {
    __shared__ __align__(16) __bf16 sA[2][128][40];
    __shared__ __align__(16) __bf16 sB[2][256][40];

    const int bm = blockIdx.x, bn = blockIdx.y;
    const int tid = threadIdx.x;
    const int wave = tid >> 5, lane = tid & 31;
    const int wm = wave & 1, wn = wave >> 1;   // 2 x 4 wave grid
    const int h = lane >> 4, ln = lane & 15;

    v8f acc[4][4] = {};

    // async-stage one 32-wide K slab into buffer `buf`
    auto stage = [&](int buf, int k0) {
        #pragma unroll
        for (int i = 0; i < 6; i++) {
            int idx = tid + i * 256;               // 512 A chunks + 1024 B chunks
            if (idx < 512) {
                int row = idx >> 2, ch = idx & 3;
                async_ld16(&sA[buf][row][ch * 8],
                           A + (size_t)(bm * 128 + row) * K + k0 + ch * 8);
            } else {
                int j = idx - 512;
                int row = j >> 2, ch = j & 3;
                async_ld16(&sB[buf][row][ch * 8],
                           B + (size_t)(bn * 256 + row) * K + k0 + ch * 8);
            }
        }
    };

    stage(0, 0);
    const int nIter = K >> 5;
    for (int it = 0; it < nIter; it++) {
        const int buf = it & 1;
        wait_async0();          // previous slab's async loads landed in LDS
        __syncthreads();        // visible to all waves; prior reads done
        if (it + 1 < nIter) stage(buf ^ 1, (it + 1) << 5);

        v16bf aF[4], bF[4];
        #pragma unroll
        for (int mi = 0; mi < 4; mi++) {
            int row = wm * 64 + mi * 16 + ln;
            v8bf lo = *(const v8bf*)&sA[buf][row][8 * h];
            v8bf hi = *(const v8bf*)&sA[buf][row][16 + 8 * h];
            #pragma unroll
            for (int e = 0; e < 8; e++) { aF[mi][e] = lo[e]; aF[mi][8 + e] = hi[e]; }
        }
        #pragma unroll
        for (int ni = 0; ni < 4; ni++) {
            int col = wn * 64 + ni * 16 + ln;
            v8bf lo = *(const v8bf*)&sB[buf][col][8 * h];
            v8bf hi = *(const v8bf*)&sB[buf][col][16 + 8 * h];
            #pragma unroll
            for (int e = 0; e < 8; e++) { bF[ni][e] = lo[e]; bF[ni][8 + e] = hi[e]; }
        }
        #pragma unroll
        for (int mi = 0; mi < 4; mi++)
            #pragma unroll
            for (int ni = 0; ni < 4; ni++)
                acc[mi][ni] = __builtin_amdgcn_wmma_f32_16x16x32_bf16(
                    false, aF[mi], false, bF[ni], (short)0, acc[mi][ni],
                    false, false);
    }

    // epilogue: C layout -> lane holds col n = ln, rows r + 8h
    #pragma unroll
    for (int mi = 0; mi < 4; mi++)
        #pragma unroll
        for (int ni = 0; ni < 4; ni++) {
            int col = bn * 256 + wn * 64 + ni * 16 + ln;
            float bv = bias ? bias[col] : 0.0f;
            #pragma unroll
            for (int r = 0; r < 8; r++) {
                int row = bm * 128 + wm * 64 + mi * 16 + r + 8 * h;
                float v = acc[mi][ni][r] + bv;
                if (OUTF32) outf[(size_t)row * N + col] = v;
                else        outb[(size_t)row * N + col] = (__bf16)v;
            }
        }
}

// ---------------------------------------------------------------------------
// RoPE + repack: qkv[S,3840] -> Qp/Kp [seg][h][L][96] (padded), Vt [seg][h][96][L]
// ---------------------------------------------------------------------------
__global__ void rope_pack_kernel(const __bf16* __restrict__ qkv,
                                 const float* __restrict__ rot,
                                 __bf16* __restrict__ Qp, __bf16* __restrict__ Kp,
                                 __bf16* __restrict__ Vt) {
    int idx = blockIdx.x * 256 + threadIdx.x;
    if (idx >= S_TOK * NH) return;
    int s = idx >> 4, hd = idx & 15;
    int seg = s >> 10, l = s & 1023;

    const __bf16* qrow = qkv + (size_t)s * NQKV + hd * HD;
    const __bf16* krow = qrow + DIM;
    const __bf16* vrow = qrow + 2 * DIM;

    size_t base = (((size_t)seg * NH + hd) * SEGL + l) * HDP;
    __bf16* qo = Qp + base;
    __bf16* ko = Kp + base;

    #pragma unroll 8
    for (int d = 0; d < 40; d++) {
        float fr = rot[(size_t)s * 40 + d];
        float c = __cosf(fr), sn = __sinf(fr);
        float q1 = (float)qrow[d], q2 = (float)qrow[d + 40];
        qo[d]      = (__bf16)(q1 * c - q2 * sn);
        qo[d + 40] = (__bf16)(q2 * c + q1 * sn);
        float k1 = (float)krow[d], k2 = (float)krow[d + 40];
        ko[d]      = (__bf16)(k1 * c - k2 * sn);
        ko[d + 40] = (__bf16)(k2 * c + k1 * sn);
    }
    for (int d = HD; d < HDP; d++) { qo[d] = (__bf16)0.0f; ko[d] = (__bf16)0.0f; }

    __bf16* vo = Vt + (((size_t)seg * NH + hd) * HDP) * SEGL + l;
    #pragma unroll 8
    for (int d = 0; d < HD; d++) vo[(size_t)d * SEGL] = vrow[d];
    for (int d = HD; d < HDP; d++) vo[(size_t)d * SEGL] = (__bf16)0.0f;
}

// ---------------------------------------------------------------------------
// Flash attention per (seg*head, 128-query block). 8 waves x 16 query rows.
// Async K/V staging; online softmax in WMMA C-register layout;
// P relayout via per-wave LDS bounce.
// ---------------------------------------------------------------------------
__global__ __launch_bounds__(256) void attn_kernel(
    const __bf16* __restrict__ Qp, const __bf16* __restrict__ Kp,
    const __bf16* __restrict__ Vt, __bf16* __restrict__ ctx) {
    __shared__ __align__(16) __bf16 sK[128][104];
    __shared__ __align__(16) __bf16 sV[96][136];
    __shared__ __align__(16) __bf16 sP[8][16][40];

    const int sh = blockIdx.x;                 // seg*16 + head
    const int tid = threadIdx.x, lane = tid & 31, wave = tid >> 5;
    const int h = lane >> 4, ln = lane & 15;

    const __bf16* Qh = Qp + (size_t)sh * SEGL * HDP;
    const __bf16* Kh = Kp + (size_t)sh * SEGL * HDP;
    const __bf16* Vh = Vt + (size_t)sh * HDP * SEGL;

    const int qBase = blockIdx.y * 128 + wave * 16;

    // Q fragments: 3 K-steps covering d = 0..95
    v16bf aQ[3];
    #pragma unroll
    for (int ks = 0; ks < 3; ks++) {
        const __bf16* qr = Qh + (size_t)(qBase + ln) * HDP + ks * 32;
        v8bf lo = *(const v8bf*)(qr + 8 * h);
        v8bf hi = *(const v8bf*)(qr + 16 + 8 * h);
        #pragma unroll
        for (int e = 0; e < 8; e++) { aQ[ks][e] = lo[e]; aQ[ks][8 + e] = hi[e]; }
    }

    v8f cacc[6] = {};
    float m8[8], l8[8];
    #pragma unroll
    for (int r = 0; r < 8; r++) { m8[r] = -1e30f; l8[r] = 0.0f; }
    const float scale = 0.1118033988749895f;   // 1/sqrt(80)

    for (int kt = 0; kt < 8; kt++) {
        // async-stage K tile [128][96] and V^T tile [96][128]
        #pragma unroll
        for (int i = 0; i < 6; i++) {
            int idx = tid + i * 256;
            int row = idx / 12, ch = idx % 12;
            async_ld16(&sK[row][ch * 8],
                       Kh + (size_t)(kt * 128 + row) * HDP + ch * 8);
        }
        #pragma unroll
        for (int i = 0; i < 6; i++) {
            int idx = tid + i * 256;
            int row = idx >> 4, ch = idx & 15;
            async_ld16(&sV[row][ch * 8],
                       Vh + (size_t)row * SEGL + kt * 128 + ch * 8);
        }
        wait_async0();
        __syncthreads();

        // scores: 16 rows x 128 keys = 8 N-tiles x 3 K-steps
        v8f sacc[8];
        #pragma unroll
        for (int nt = 0; nt < 8; nt++) {
            v8f a = {};
            #pragma unroll
            for (int ks = 0; ks < 3; ks++) {
                int col = nt * 16 + ln;
                v8bf lo = *(const v8bf*)&sK[col][ks * 32 + 8 * h];
                v8bf hi = *(const v8bf*)&sK[col][ks * 32 + 16 + 8 * h];
                v16bf bK;
                #pragma unroll
                for (int e = 0; e < 8; e++) { bK[e] = lo[e]; bK[8 + e] = hi[e]; }
                a = __builtin_amdgcn_wmma_f32_16x16x32_bf16(
                    false, aQ[ks], false, bK, (short)0, a, false, false);
            }
            #pragma unroll
            for (int r = 0; r < 8; r++) a[r] *= scale;
            sacc[nt] = a;
        }

        // online softmax row stats (row m = r + 8h lives across 16 lanes)
        #pragma unroll
        for (int r = 0; r < 8; r++) {
            float lm = sacc[0][r];
            #pragma unroll
            for (int nt = 1; nt < 8; nt++) lm = fmaxf(lm, sacc[nt][r]);
            lm = fmaxf(lm, __shfl_xor(lm, 1, 32));
            lm = fmaxf(lm, __shfl_xor(lm, 2, 32));
            lm = fmaxf(lm, __shfl_xor(lm, 4, 32));
            lm = fmaxf(lm, __shfl_xor(lm, 8, 32));
            float nm = fmaxf(m8[r], lm);
            float sc = __expf(m8[r] - nm);
            l8[r] *= sc;
            #pragma unroll
            for (int dt = 0; dt < 6; dt++) cacc[dt][r] *= sc;
            m8[r] = nm;
        }
        #pragma unroll
        for (int nt = 0; nt < 8; nt++)
            #pragma unroll
            for (int r = 0; r < 8; r++)
                sacc[nt][r] = __expf(sacc[nt][r] - m8[r]);
        #pragma unroll
        for (int r = 0; r < 8; r++) {
            float ls = 0.0f;
            #pragma unroll
            for (int nt = 0; nt < 8; nt++) ls += sacc[nt][r];
            ls += __shfl_xor(ls, 1, 32);
            ls += __shfl_xor(ls, 2, 32);
            ls += __shfl_xor(ls, 4, 32);
            ls += __shfl_xor(ls, 8, 32);
            l8[r] += ls;
        }

        // P @ V in 32-key groups: C->A relayout via per-wave LDS bounce
        #pragma unroll
        for (int j = 0; j < 4; j++) {
            #pragma unroll
            for (int half2 = 0; half2 < 2; half2++) {
                int nt = 2 * j + half2;
                #pragma unroll
                for (int r = 0; r < 8; r++)
                    sP[wave][r + 8 * h][half2 * 16 + ln] = (__bf16)sacc[nt][r];
            }
            asm volatile("s_wait_dscnt 0" ::: "memory");
            v8bf lo = *(const v8bf*)&sP[wave][ln][8 * h];
            v8bf hi = *(const v8bf*)&sP[wave][ln][16 + 8 * h];
            v16bf aP;
            #pragma unroll
            for (int e = 0; e < 8; e++) { aP[e] = lo[e]; aP[8 + e] = hi[e]; }
            #pragma unroll
            for (int dt = 0; dt < 6; dt++) {
                int dcol = dt * 16 + ln;
                v8bf blo = *(const v8bf*)&sV[dcol][j * 32 + 8 * h];
                v8bf bhi = *(const v8bf*)&sV[dcol][j * 32 + 16 + 8 * h];
                v16bf bV;
                #pragma unroll
                for (int e = 0; e < 8; e++) { bV[e] = blo[e]; bV[8 + e] = bhi[e]; }
                cacc[dt] = __builtin_amdgcn_wmma_f32_16x16x32_bf16(
                    false, aP, false, bV, (short)0, cacc[dt], false, false);
            }
        }
        __syncthreads();
    }

    // finalize: ctx[s, head*80 + d] (only d < 80 valid)
    const int seg = sh >> 4, head = sh & 15;
    #pragma unroll
    for (int r = 0; r < 8; r++) {
        float inv = 1.0f / l8[r];
        int row = seg * SEGL + blockIdx.y * 128 + wave * 16 + r + 8 * h;
        #pragma unroll
        for (int dt = 0; dt < 5; dt++) {
            int d = dt * 16 + ln;
            ctx[(size_t)row * DIM + head * HD + d] = (__bf16)(cacc[dt][r] * inv);
        }
    }
}

// ---------------------------------------------------------------------------
// launch
// ---------------------------------------------------------------------------
extern "C" void kernel_launch(void* const* d_in, const int* in_sizes, int n_in,
                              void* d_out, int out_size, void* d_ws, size_t ws_size,
                              hipStream_t stream) {
    const float* hidden = (const float*)d_in[0];
    const float* rot    = (const float*)d_in[1];
    const float* w_qkv  = (const float*)d_in[2];
    const float* b_qkv  = (const float*)d_in[3];
    const float* w_proj = (const float*)d_in[4];
    const float* b_proj = (const float*)d_in[5];
    float* out = (float*)d_out;

    char* ws = (char*)d_ws;
    size_t off = 0;
    __bf16* Xb     = (__bf16*)(ws + off); off += (size_t)S_TOK * DIM * 2;
    __bf16* Wqkvb  = (__bf16*)(ws + off); off += (size_t)NQKV * DIM * 2;
    __bf16* Wprojb = (__bf16*)(ws + off); off += (size_t)DIM * DIM * 2;
    __bf16* QKVb   = (__bf16*)(ws + off); off += (size_t)S_TOK * NQKV * 2;
    __bf16* Qp     = (__bf16*)(ws + off); off += (size_t)NSEG * NH * SEGL * HDP * 2;
    __bf16* Kp     = (__bf16*)(ws + off); off += (size_t)NSEG * NH * SEGL * HDP * 2;
    __bf16* Vt     = (__bf16*)(ws + off); off += (size_t)NSEG * NH * HDP * SEGL * 2;
    __bf16* Ctxb   = (__bf16*)(ws + off); off += (size_t)S_TOK * DIM * 2;

    // 1. conversions
    cvt_bf16_kernel<<<4096, 256, 0, stream>>>(hidden, Xb, S_TOK * DIM);
    cvt_bf16_kernel<<<4096, 256, 0, stream>>>(w_qkv, Wqkvb, NQKV * DIM);
    cvt_bf16_kernel<<<2048, 256, 0, stream>>>(w_proj, Wprojb, DIM * DIM);

    // 2. QKV projection: [8192,1280] @ [3840,1280]^T + b_qkv -> bf16
    gemm_bf16_kernel<0><<<dim3(S_TOK / 128, NQKV / 256), 256, 0, stream>>>(
        Xb, Wqkvb, b_qkv, QKVb, nullptr, S_TOK, NQKV, DIM);

    // 3. RoPE + repack into attention layouts
    rope_pack_kernel<<<(S_TOK * NH) / 256, 256, 0, stream>>>(QKVb, rot, Qp, Kp, Vt);

    // 4. block-diagonal flash attention
    attn_kernel<<<dim3(NSEG * NH, SEGL / 128), 256, 0, stream>>>(Qp, Kp, Vt, Ctxb);

    // 5. output projection: [8192,1280] @ [1280,1280]^T + b_proj -> f32 out
    gemm_bf16_kernel<1><<<dim3(S_TOK / 128, DIM / 128 / 2), 256, 0, stream>>>(
        Ctxb, Wprojb, b_proj, nullptr, out, S_TOK, DIM, DIM);
}
